// LlamaAttention_25469156065903
// MI455X (gfx1250) — compile-verified, matrix operand order checked by reading, not run on hardware
//
#include <hip/hip_runtime.h>
#include <hip/hip_bf16.h>

// ---------------------------------------------------------------------------
// Llama attention layer for MI455X (gfx1250, wave32, WMMA).
// - All matmuls: v_wmma_f32_16x16x32_bf16 (f32 accum, bf16 operands).
// - KV cache stored pre-swizzled in WMMA B-fragment image order, so attention
//   staging is a raw 16KB copy done by the Tensor Data Mover (TDM).
// - GEMM templated on (N, K) so epilogue/staging addresses fold to immediates.
// Workspace use: ~144 MB.
// ---------------------------------------------------------------------------

#define NHEADS 32
#define NKVH   8
#define HDIM   128
#define BATCH  4
#define SEQ    1024
#define HID    4096
#define KVBLK  64
#define BPSEQ  16           // SEQ / KVBLK
#define TOKENS 4096         // BATCH * SEQ
#define QSCALE 0.08838834764831845f   // 1/sqrt(128)
#define KVIMG  8192         // bhalf elems per (block, kv head) fragment image

typedef __bf16 bhalf;
typedef __attribute__((ext_vector_type(16))) bhalf v16bf;
typedef __attribute__((ext_vector_type(4)))  bhalf v4bf;
typedef __attribute__((ext_vector_type(8)))  float v8f;
typedef __attribute__((ext_vector_type(4)))  float v4f;
typedef __attribute__((ext_vector_type(4)))  unsigned int u32x4;
typedef __attribute__((ext_vector_type(8)))  int i32x8;
typedef __attribute__((ext_vector_type(4)))  int i32x4;

union FragU { v16bf v; unsigned int u[8]; };

static __device__ inline v8f zero_v8f() {
    v8f z;
    #pragma unroll
    for (int i = 0; i < 8; ++i) z[i] = 0.0f;
    return z;
}

// Fragment-layout helpers (CDNA5 ISA 7.12.2, 16-bit 16x32 A / 32x16 B):
// lane = idx%16 + c*16, c = (k>>3)&1 picks {0..7,16..23} vs {8..15,24..31};
// element e = k' (<8) or k'-8 (k' in 16..23), k' = k - 8c.
static __device__ inline int frag_c(int k)  { return (k >> 3) & 1; }
static __device__ inline int frag_e(int k)  {
    int kp = k - 8 * ((k >> 3) & 1);
    return (kp < 8) ? kp : kp - 8;
}

// ---------------------------------------------------------------------------
// TDM: contiguous 16KB global->LDS copy (1-D tensor descriptor, 8B elements).
// D# per CDNA5 ISA. Tracked by TENSORcnt.
// 6-arg builtin form (clang-23 / therock-10.0 headers).
// ---------------------------------------------------------------------------
static __device__ inline void tdm_copy_16kb(const bhalf* gsrc, unsigned lds_off)
{
    unsigned long long ga = (unsigned long long)(uintptr_t)gsrc;
    u32x4 g0;
    g0[0] = 1u;                                   // count=1, user mode, no gather
    g0[1] = lds_off;                              // lds_addr (bytes)
    g0[2] = (unsigned)ga;                         // global_addr[31:0]
    g0[3] = (unsigned)((ga >> 32) & 0x01FFFFFFu)  // global_addr[56:32]
            | (2u << 30);                         // type = 2 ("image")
    i32x8 g1;
    g1[0] = (3 << 16);                            // wg_mask=0, data_size=8B
    g1[1] = (int)((2048u & 0xFFFFu) << 16);       // tensor_dim0[15:0] = 2048
    g1[2] = (int)((2048u >> 16) | (1u << 16));    // tensor_dim0 hi | tensor_dim1=1
    g1[3] = (int)(2048u << 16);                   // tensor_dim1 hi=0 | tile_dim0=2048
    g1[4] = 1;                                    // tile_dim1=1, tile_dim2=0
    g1[5] = 2048;                                 // tensor_dim0_stride = 2048
    g1[6] = 0;
    g1[7] = 0;
    i32x4 gz4 = {0, 0, 0, 0};
    i32x8 gz8 = {0, 0, 0, 0, 0, 0, 0, 0};
    __builtin_amdgcn_tensor_load_to_lds(g0, g1, gz4, gz4, gz8, 0);
}

// ---------------------------------------------------------------------------
// Generic f32-in / f32-out GEMM, bf16 WMMA compute.  C[M,N] = A[M,K] * B[K,N].
// N, K compile-time so all addressing folds into immediate offsets.
// 256 threads, 128x128 tile, each wave computes 64x32 (4x2 WMMA tiles).
// ---------------------------------------------------------------------------
template <int N, int K>
__global__ __launch_bounds__(256)
void gemm_xw_bf16(const float* __restrict__ A, const float* __restrict__ Bw,
                  float* __restrict__ C)
{
    __shared__ bhalf As[128 * 32];   // [m-block 8][lane 32][elem 16]
    __shared__ bhalf Bs[32 * 128];   // [n-block 8][lane 32][elem 16]

    const int tid  = threadIdx.x;
    const int lane = tid & 31;
    const int wave = tid >> 5;
    const int bm = blockIdx.y * 128;
    const int bn = blockIdx.x * 128;
    const int wr = wave >> 2;        // 0..1 : 64-row band
    const int wc = wave & 3;         // 0..3 : 32-col band

    v8f acc[4][2];
    #pragma unroll
    for (int i = 0; i < 4; ++i)
        #pragma unroll
        for (int j = 0; j < 2; ++j) acc[i][j] = zero_v8f();

    // A staging: thread = (row 0..127, c 0..1); one 32B fragment row per thread.
    const int a_row = tid >> 1;
    const int a_c   = tid & 1;
    bhalf* a_dst = &As[(a_row >> 4) * 512 + ((a_row & 15) + a_c * 16) * 16];
    const float* a_src = A + (size_t)(bm + a_row) * K + a_c * 8;
    // B staging: thread = (k-quad 0..7, n-quad 0..31); 4x b64 stores per thread.
    const int b_k4 = (tid >> 5) * 4;       // 0,4,...,28
    const int b_n4 = (tid & 31) * 4;       // 0..124
    const int b_c  = frag_c(b_k4);
    const int b_e  = frag_e(b_k4);
    const float* b_src = Bw + (size_t)b_k4 * N + bn + b_n4;

    for (int k0 = 0; k0 < K; k0 += 32) {
        __syncthreads();
        {   // stage A tile 128x32 -> A-fragment order (one v16bf per thread)
            const float* src = a_src + k0;
            v4f f0 = *(const v4f*)(src);
            v4f f1 = *(const v4f*)(src + 4);
            v4f f2 = *(const v4f*)(src + 16);
            v4f f3 = *(const v4f*)(src + 20);
            v16bf pk;
            #pragma unroll
            for (int i = 0; i < 4; ++i) {
                pk[i]      = (bhalf)f0[i];
                pk[4 + i]  = (bhalf)f1[i];
                pk[8 + i]  = (bhalf)f2[i];
                pk[12 + i] = (bhalf)f3[i];
            }
            *(v16bf*)a_dst = pk;
        }
        {   // stage B tile 32x128 -> B-fragment order (4x v4bf per thread)
            const float* src = b_src + (size_t)k0 * N;
            v4f fr[4];
            #pragma unroll
            for (int kk = 0; kk < 4; ++kk)
                fr[kk] = *(const v4f*)(src + (size_t)kk * N);
            #pragma unroll
            for (int nn = 0; nn < 4; ++nn) {
                int n = b_n4 + nn;
                v4bf pk;
                #pragma unroll
                for (int kk = 0; kk < 4; ++kk) pk[kk] = (bhalf)fr[kk][nn];
                *(v4bf*)&Bs[(n >> 4) * 512 + ((n & 15) + b_c * 16) * 16 + b_e] = pk;
            }
        }
        __syncthreads();

        v16bf afr[4], bfr[2];
        #pragma unroll
        for (int i = 0; i < 4; ++i)
            afr[i] = *(const v16bf*)&As[(wr * 4 + i) * 512 + lane * 16];
        #pragma unroll
        for (int j = 0; j < 2; ++j)
            bfr[j] = *(const v16bf*)&Bs[(wc * 2 + j) * 512 + lane * 16];

        #pragma unroll
        for (int i = 0; i < 4; ++i)
            #pragma unroll
            for (int j = 0; j < 2; ++j)
                acc[i][j] = __builtin_amdgcn_wmma_f32_16x16x32_bf16(
                    false, afr[i], false, bfr[j], (short)0, acc[i][j], false, false);
    }

    // epilogue: single base pointer, all offsets compile-time immediates
    const int hf = lane >> 4;
    const int cn = lane & 15;
    float* cbase = C + (size_t)(bm + wr * 64 + hf * 8) * N + (bn + wc * 32 + cn);
    #pragma unroll
    for (int i = 0; i < 4; ++i)
        #pragma unroll
        for (int j = 0; j < 2; ++j) {
            #pragma unroll
            for (int r = 0; r < 8; ++r)
                cbase[(i * 16 + r) * N + j * 16] = acc[i][j][r];
        }
}

// ---------------------------------------------------------------------------
// RoPE (+ 1/sqrt(d) folded into Q), bf16 convert; K/V written straight into
// the paged cache in WMMA B-fragment image order (16KB per block per kv head).
// grid (TOKENS, 48), block 64.  y<32: Q head; y<40: K head; else V head.
// ---------------------------------------------------------------------------
__global__ __launch_bounds__(64)
void rope_fill(const float* __restrict__ qf, const float* __restrict__ kf,
               const float* __restrict__ vf, const float* __restrict__ inv_freq,
               const int* __restrict__ pos_ids, const int* __restrict__ block_off,
               bhalf* __restrict__ qb, bhalf* __restrict__ kcache,
               bhalf* __restrict__ vcache)
{
    const int t  = blockIdx.x;           // global token
    const int hh = blockIdx.y;           // 0..47
    const int d  = threadIdx.x;          // rotation pair 0..63
    const int b  = t / SEQ;
    const int s  = t - b * SEQ;

    const float fr = (float)pos_ids[t] * inv_freq[d];
    const float cn = cosf(fr);
    const float sn = sinf(fr);

    if (hh < NHEADS) {
        const float* src = qf + (size_t)t * (NHEADS * HDIM) + hh * HDIM;
        float x1 = src[d], x2 = src[d + 64];
        bhalf* dst = qb + (((size_t)b * NHEADS + hh) * SEQ + s) * HDIM;
        dst[d]      = (bhalf)((x1 * cn - x2 * sn) * QSCALE);
        dst[d + 64] = (bhalf)((x2 * cn + x1 * sn) * QSCALE);
    } else {
        const int lb   = s >> 6;
        const int phys = block_off[b * BPSEQ + lb];
        const int rin  = s & 63;                     // kv row within block
        if (hh < NHEADS + NKVH) {
            // K image: [nb4][ko4][lane32][e16]; nb/n from kv row, ko/c/e from dim
            const int hk = hh - NHEADS;
            const float* src = kf + (size_t)t * (NKVH * HDIM) + hk * HDIM;
            bhalf* img = kcache + ((size_t)phys * NKVH + hk) * KVIMG;
            float x1 = src[d], x2 = src[d + 64];
            float o1 = x1 * cn - x2 * sn;
            float o2 = x2 * cn + x1 * sn;
            const int nb = rin >> 4, n = rin & 15;
            #pragma unroll
            for (int p = 0; p < 2; ++p) {
                int dd = d + p * 64;
                int dl = dd & 31;
                img[nb * 2048 + (dd >> 5) * 512 +
                    (n + frag_c(dl) * 16) * 16 + frag_e(dl)] = (bhalf)(p ? o2 : o1);
            }
        } else {
            // V image: [nbd8][ko2][lane32][e16]; nbd/n from dim, ko/c/e from kv row
            const int hv = hh - NHEADS - NKVH;
            const float* src = vf + (size_t)t * (NKVH * HDIM) + hv * HDIM;
            bhalf* img = vcache + ((size_t)phys * NKVH + hv) * KVIMG;
            const int ko2 = rin >> 5;
            const int cv = frag_c(rin & 31), ev = frag_e(rin & 31);
            #pragma unroll
            for (int p = 0; p < 2; ++p) {
                int dd = d + p * 64;
                img[(dd >> 4) * 1024 + ko2 * 512 +
                    ((dd & 15) + cv * 16) * 16 + ev] = (bhalf)src[dd];
            }
        }
    }
}

// ---------------------------------------------------------------------------
// Flash-style causal paged attention (GQA 32q/8kv heads, HD=128).
// grid (SEQ/64, NHEADS, BATCH), block 128 (4 waves; wave = 16 query rows).
// K/V chunks staged by the Tensor Data Mover (cache already fragment-ordered).
// ---------------------------------------------------------------------------
__global__ __launch_bounds__(128)
void paged_attn(const bhalf* __restrict__ qb, const bhalf* __restrict__ kc,
                const bhalf* __restrict__ vc, const int* __restrict__ block_off,
                float* __restrict__ out)
{
    __shared__ bhalf Ks[4 * 2048];   // K chunk, B-frag order [nb4][ko4][lane32][e16]
    __shared__ bhalf Vs[8 * 1024];   // V chunk, B-frag order [nbd8][ko2][lane32][e16]
    __shared__ bhalf Ps[4][1024];    // per-wave P tile, A-frag order [ko2][lane32][e16]

    const int tid  = threadIdx.x;
    const int lane = tid & 31;
    const int wave = tid >> 5;
    const int qbk  = blockIdx.x;     // query 64-block
    const int h    = blockIdx.y;
    const int b    = blockIdx.z;
    const int kvh  = h >> 2;         // GQA: 4 q heads per kv head
    const int hf   = lane >> 4;      // C-layout row half
    const int cn   = lane & 15;

    const unsigned ks_off = (unsigned)(uintptr_t)(&Ks[0]);
    const unsigned vs_off = (unsigned)(uintptr_t)(&Vs[0]);

    // --- resident Q fragments (A layout), loaded as aligned dword pairs ---
    const int m = lane & 15, cq = lane >> 4;
    const unsigned int* qrow = (const unsigned int*)
        (qb + (((size_t)b * NHEADS + h) * SEQ + qbk * 64 + wave * 16 + m) * HDIM);
    v16bf qfr[4];
    #pragma unroll
    for (int ko = 0; ko < 4; ++ko) {
        FragU fu;
        #pragma unroll
        for (int v = 0; v < 8; ++v) {
            int kpair = ko * 32 + cq * 8 + (v < 4 ? 2 * v : 2 * v + 8);
            fu.u[v] = qrow[kpair >> 1];
        }
        qfr[ko] = fu.v;
    }

    v8f oacc[8];
    #pragma unroll
    for (int i = 0; i < 8; ++i) oacc[i] = zero_v8f();
    float m_r[8], l_r[8];
    #pragma unroll
    for (int r = 0; r < 8; ++r) { m_r[r] = -1e30f; l_r[r] = 0.0f; }

    for (int j = 0; j <= qbk; ++j) {
        __syncthreads();                       // all waves done with prev chunk
        if (wave == 0) {
            const int phys = block_off[b * BPSEQ + j];
            const bhalf* kimg = kc + ((size_t)phys * NKVH + kvh) * KVIMG;
            const bhalf* vimg = vc + ((size_t)phys * NKVH + kvh) * KVIMG;
            tdm_copy_16kb(kimg, ks_off);
            tdm_copy_16kb(vimg, vs_off);
            __builtin_amdgcn_s_wait_tensorcnt(0);
        }
        __syncthreads();                       // LDS images visible to all waves

        // --- scores: S[16x64] = Q[16x128] * K^T ---
        v8f sacc[4];
        #pragma unroll
        for (int nb = 0; nb < 4; ++nb) sacc[nb] = zero_v8f();
        #pragma unroll
        for (int nb = 0; nb < 4; ++nb)
            #pragma unroll
            for (int ko = 0; ko < 4; ++ko) {
                v16bf kfr = *(const v16bf*)&Ks[nb * 2048 + ko * 512 + lane * 16];
                sacc[nb] = __builtin_amdgcn_wmma_f32_16x16x32_bf16(
                    false, qfr[ko], false, kfr, (short)0, sacc[nb], false, false);
            }

        // causal mask: only the diagonal chunk needs it
        if (j == qbk) {
            #pragma unroll
            for (int nb = 0; nb < 4; ++nb) {
                int col = nb * 16 + cn;
                #pragma unroll
                for (int r = 0; r < 8; ++r)
                    if (col > wave * 16 + r + 8 * hf) sacc[nb][r] = -1e30f;
            }
        }

        // --- online softmax (row stats reduced over 16-lane halves) ---
        float scl[8];
        #pragma unroll
        for (int r = 0; r < 8; ++r) {
            float mx = fmaxf(fmaxf(sacc[0][r], sacc[1][r]),
                             fmaxf(sacc[2][r], sacc[3][r]));
            #pragma unroll
            for (int dlt = 1; dlt < 16; dlt <<= 1)
                mx = fmaxf(mx, __shfl_xor(mx, dlt, 32));
            float nm = fmaxf(m_r[r], mx);
            scl[r] = __expf(m_r[r] - nm);
            m_r[r] = nm;
            float ps = 0.0f;
            #pragma unroll
            for (int nb = 0; nb < 4; ++nb) {
                float p = __expf(sacc[nb][r] - nm);
                sacc[nb][r] = p;
                ps += p;
            }
            #pragma unroll
            for (int dlt = 1; dlt < 16; dlt <<= 1)
                ps += __shfl_xor(ps, dlt, 32);
            l_r[r] = l_r[r] * scl[r] + ps;
        }
        #pragma unroll
        for (int nbd = 0; nbd < 8; ++nbd)
            #pragma unroll
            for (int r = 0; r < 8; ++r) oacc[nbd][r] *= scl[r];

        // --- C-layout -> A-layout transpose of P through per-wave LDS ---
        #pragma unroll
        for (int nb = 0; nb < 4; ++nb) {
            int colk = nb * 16 + cn;                 // K index 0..63
            int ko = colk >> 5, kl = colk & 31;
            int cc = frag_c(kl), e = frag_e(kl);
            #pragma unroll
            for (int r = 0; r < 8; ++r) {
                int mm = r + 8 * hf;
                Ps[wave][ko * 512 + (mm + cc * 16) * 16 + e] = (bhalf)sacc[nb][r];
            }
        }
        __builtin_amdgcn_fence(__ATOMIC_ACQ_REL, "wavefront");  // order wave-local LDS

        v16bf pfr[2];
        pfr[0] = *(const v16bf*)&Ps[wave][lane * 16];
        pfr[1] = *(const v16bf*)&Ps[wave][512 + lane * 16];

        // --- O += P[16x64] * V[64x128] ---
        #pragma unroll
        for (int nbd = 0; nbd < 8; ++nbd)
            #pragma unroll
            for (int ko = 0; ko < 2; ++ko) {
                v16bf vfr = *(const v16bf*)&Vs[nbd * 1024 + ko * 512 + lane * 16];
                oacc[nbd] = __builtin_amdgcn_wmma_f32_16x16x32_bf16(
                    false, pfr[ko], false, vfr, (short)0, oacc[nbd], false, false);
            }
    }

    // --- epilogue: normalize, write attn[token, h*128 + d] (f32) ---
    float* dst0 = out + ((size_t)b * SEQ + qbk * 64 + wave * 16 + 8 * hf)
                        * (NHEADS * HDIM) + h * HDIM + cn;
    #pragma unroll
    for (int r = 0; r < 8; ++r) {
        float inv = 1.0f / l_r[r];
        #pragma unroll
        for (int nbd = 0; nbd < 8; ++nbd)
            dst0[r * (NHEADS * HDIM) + nbd * 16] = oacc[nbd][r] * inv;
    }
}

// ---------------------------------------------------------------------------
extern "C" void kernel_launch(void* const* d_in, const int* in_sizes, int n_in,
                              void* d_out, int out_size, void* d_ws, size_t ws_size,
                              hipStream_t stream)
{
    const float* X        = (const float*)d_in[0];
    const float* Wq       = (const float*)d_in[1];
    const float* Wk       = (const float*)d_in[2];
    const float* Wv       = (const float*)d_in[3];
    const float* Wo       = (const float*)d_in[4];
    const float* inv_freq = (const float*)d_in[5];
    // d_in[6]/d_in[7] (f32 zero caches) unused: bf16 cache lives in workspace.
    const int* pos  = (const int*)d_in[8];
    const int* boff = (const int*)d_in[9];
    float* out = (float*)d_out;

    char* ws = (char*)d_ws;                           // needs ~144 MB
    float* qf32   = (float*)(ws);                     // 64 MB, reused as attn out
    float* kf32   = (float*)(ws + (64ull  << 20));    // 16 MB
    float* vf32   = (float*)(ws + (80ull  << 20));    // 16 MB
    bhalf* qbf    = (bhalf*)(ws + (96ull  << 20));    // 32 MB  [B,NH,S,HD]
    bhalf* kcache = (bhalf*)(ws + (128ull << 20));    //  8 MB  fragment images
    bhalf* vcache = (bhalf*)(ws + (136ull << 20));    //  8 MB  fragment images

    gemm_xw_bf16<HID, HID><<<dim3(HID / 128, TOKENS / 128), dim3(256), 0, stream>>>(
        X, Wq, qf32);
    gemm_xw_bf16<NKVH * HDIM, HID>
        <<<dim3((NKVH * HDIM) / 128, TOKENS / 128), dim3(256), 0, stream>>>(
        X, Wk, kf32);
    gemm_xw_bf16<NKVH * HDIM, HID>
        <<<dim3((NKVH * HDIM) / 128, TOKENS / 128), dim3(256), 0, stream>>>(
        X, Wv, vf32);

    rope_fill<<<dim3(TOKENS, 48), dim3(64), 0, stream>>>(
        qf32, kf32, vf32, inv_freq, pos, boff, qbf, kcache, vcache);

    paged_attn<<<dim3(SEQ / 64, NHEADS, BATCH), dim3(128), 0, stream>>>(
        qbf, kcache, vcache, boff, qf32 /* attn out, reuses Q buffer */);

    gemm_xw_bf16<HID, HID><<<dim3(HID / 128, TOKENS / 128), dim3(256), 0, stream>>>(
        qf32, Wo, out);
}
